// SelfAttentionSparse_86535001079844
// MI455X (gfx1250) — matrix operand order repaired and checked by reading, not compiled, not run on hardware
//
#include <hip/hip_runtime.h>
#include <hip/hip_bf16.h>

#define DEV __device__ __forceinline__

typedef __attribute__((ext_vector_type(16))) __bf16 v16bf;
typedef __attribute__((ext_vector_type(8)))  float  v8f;

union FragU {
  v16bf v;
  unsigned u[8];
};

// Problem sizes (fixed by the reference)
constexpr int B = 4, C = 256, L = 2048, H = 8, E = 256;
constexpr float INV_SQRT_C = 0.0625f;  // 1/sqrt(256)

// ---------------------------------------------------------------------------
// CDNA5 async global->LDS copies (ASYNCcnt path). Probe result: the builtin
// exists with 4 args and generic `int*` parameters. Async-to-LDS takes a
// per-lane LDS destination, so it performs the fragment-major swizzle inside
// the DMA itself (no VGPR round-trip).
// ---------------------------------------------------------------------------
#if defined(__AMDGCN__) && __has_builtin(__builtin_amdgcn_global_load_async_to_lds_b32)
#define HAVE_ASYNC_LDS 1
#else
#define HAVE_ASYNC_LDS 0
#endif

#if HAVE_ASYNC_LDS
DEV void async_cp_b32(void* lds_generic, const void* g) {
  __builtin_amdgcn_global_load_async_to_lds_b32(
      (int*)g, (int*)lds_generic, /*offset=*/0, /*cpol=*/0);
}
#endif

DEV void wait_async() {
#if HAVE_ASYNC_LDS
#if __has_builtin(__builtin_amdgcn_s_wait_asynccnt)
  __builtin_amdgcn_s_wait_asynccnt(0);
#else
  asm volatile("s_wait_asynccnt 0x0" ::: "memory");
#endif
#endif
}

DEV __bf16 f2bf(float f) {
  unsigned u = __builtin_bit_cast(unsigned, f);
  unsigned r = u + 0x7FFFu + ((u >> 16) & 1u);   // round-to-nearest-even
  unsigned short hbits = (unsigned short)(r >> 16);
  return __builtin_bit_cast(__bf16, hbits);
}

DEV unsigned pack2(float a, float b) {
  unsigned short x = __builtin_bit_cast(unsigned short, f2bf(a));
  unsigned short y = __builtin_bit_cast(unsigned short, f2bf(b));
  return (unsigned)x | ((unsigned)y << 16);
}

DEV v8f wmma_bf16(v16bf a, v16bf b, v8f c) {
  return __builtin_amdgcn_wmma_f32_16x16x32_bf16(
      false, a, false, b, (short)0, c, false, false);
}

DEV void zero8(v8f& v) {
#pragma unroll
  for (int k = 0; k < 8; ++k) v[k] = 0.0f;
}

// ---------------------------------------------------------------------------
// Fragment-major LDS staging (per ISA 7.12.2 register images).
// A[m][k] slot: lane = (m&15) + 16*((k>>3)&1), elem = (k&7)+8*(k>>4); element
// pairs are contiguous K -> one u32 global load == one u32 LDS slot.
// B[k][n] slot: lane = (n&15) + 16*((k>>3)&1), elem = (k&7)+8*(k>>4).
// ---------------------------------------------------------------------------

// Stage BM x 32 row-major bf16 A tile -> fragment-major (BM/16 subtiles).
template <int BM>
DEV void stage_a_bf16(v16bf* lds, const __bf16* gA, int lda, int tid) {
  unsigned* du = (unsigned*)lds;
  for (int idx = tid; idx < BM * 16; idx += 256) {
    int m = idx >> 4;
    int k = (idx & 15) * 2;
    int lane = (m & 15) + 16 * ((k >> 3) & 1);
    int elem = (k & 7) + ((k >> 4) << 3);
    unsigned* dst = &du[((m >> 4) * 32 + lane) * 8 + (elem >> 1)];
    const void* src = gA + (size_t)m * lda + k;
#if HAVE_ASYNC_LDS
    async_cp_b32(dst, src);           // DMA straight into the fragment slot
#else
    *dst = *(const unsigned*)src;
#endif
  }
}

// Same but fp32 source (converted to bf16 in flight; VGPR path only).
template <int BM>
DEV void stage_a_f32(v16bf* lds, const float* gA, int lda, int tid) {
  unsigned* du = (unsigned*)lds;
  for (int idx = tid; idx < BM * 16; idx += 256) {
    int m = idx >> 4;
    int k = (idx & 15) * 2;
    const float2 f = *(const float2*)(gA + (size_t)m * lda + k);
    int lane = (m & 15) + 16 * ((k >> 3) & 1);
    int elem = (k & 7) + ((k >> 4) << 3);
    du[((m >> 4) * 32 + lane) * 8 + (elem >> 1)] = pack2(f.x, f.y);
  }
}

// Stage 32 x BN row-major bf16 B tile -> fragment-major (BN/16 subtiles).
template <int BN, int NPSHIFT>  // NPSHIFT = log2(BN/2)
DEV void stage_b_bf16(v16bf* lds, const __bf16* gB, int ldb, int tid) {
  __bf16* db = (__bf16*)lds;
  for (int idx = tid; idx < 16 * BN; idx += 256) {
    int k = idx >> NPSHIFT;
    int n = (idx & ((BN / 2) - 1)) * 2;
    unsigned val = *(const unsigned*)(gB + (size_t)k * ldb + n);
    int base = (((n >> 4) * 32) + (n & 15) + 16 * ((k >> 3) & 1)) * 16 + (k & 7) + ((k >> 4) << 3);
    db[base]      = __builtin_bit_cast(__bf16, (unsigned short)(val & 0xffff));
    db[base + 16] = __builtin_bit_cast(__bf16, (unsigned short)(val >> 16));
  }
}

// Direct-from-global A fragment (row-major, u32 pair loads, L2-resident data).
DEV v16bf load_a_frag_g(const __bf16* A, int lda, int lane) {
  int m = lane & 15, hi = lane >> 4;
  const unsigned* row = (const unsigned*)(A + (size_t)m * lda);
  FragU f;
#pragma unroll
  for (int j = 0; j < 8; ++j) {
    int kb = ((j < 4) ? (2 * j) : (16 + 2 * (j - 4))) + hi * 8;
    f.u[j] = row[kb >> 1];
  }
  return f.v;
}

// ---------------------------------------------------------------------------
// Kernel 1: x -> bf16 ; unifyheads [h][cout][t] -> Ut [cout][h][t] bf16
// ---------------------------------------------------------------------------
__global__ void prep_kernel(const float* __restrict__ x,
                            const float* __restrict__ unify,
                            __bf16* __restrict__ xbf,
                            __bf16* __restrict__ Ut) {
  int stride = gridDim.x * blockDim.x;
  int base = blockIdx.x * blockDim.x + threadIdx.x;
  const int nx = (B * C * L) / 2;
  unsigned* xo = (unsigned*)xbf;
  for (int idx = base; idx < nx; idx += stride) {
    const float2 f = *(const float2*)(x + (size_t)idx * 2);
    xo[idx] = pack2(f.x, f.y);
  }
  const int nu = C * H * C;
  for (int idx = base; idx < nu; idx += stride) {
    int t  = idx & (C - 1);
    int h  = (idx >> 8) & (H - 1);
    int co = idx >> 11;
    Ut[idx] = f2bf(unify[((size_t)h * C + co) * C + t]);
  }
}

// ---------------------------------------------------------------------------
// Kernel 2: per (b,h): Kt[l][e] = (tokey[h] @ x[b])^T, Q[e][l] = toquery[h]@x[b]
// ---------------------------------------------------------------------------
__global__ __launch_bounds__(256)
void kq_gemm_kernel(const float* __restrict__ tokey,
                    const float* __restrict__ toquery,
                    const __bf16* __restrict__ xbf,
                    __bf16* __restrict__ Kt,   // [B*H][L][E]
                    __bf16* __restrict__ Qb) { // [B*H][E][L]
  int z = blockIdx.z;
  int which = z & 1;
  int bh = z >> 1;
  int b = bh / H, h = bh - b * H;
  int e0 = blockIdx.y * 64;
  int l0 = blockIdx.x * 128;

  const float* W = (which ? toquery : tokey) + (size_t)h * E * C;
  const __bf16* Xb = xbf + (size_t)b * C * L;

  __shared__ v16bf AsF[4 * 32];   // 4 KB
  __shared__ v16bf BsF[8 * 32];   // 8 KB

  int tid = threadIdx.x, lane = tid & 31, wid = tid >> 5;
  int wm = (wid >> 2) * 2;
  int wn = (wid & 3) * 2;

  v8f acc[2][2];
#pragma unroll
  for (int i = 0; i < 2; ++i)
#pragma unroll
    for (int j = 0; j < 2; ++j) zero8(acc[i][j]);

  for (int k0 = 0; k0 < C; k0 += 32) {
    stage_a_f32<64>(AsF, W + (size_t)e0 * C + k0, C, tid);
    stage_b_bf16<128, 6>(BsF, Xb + (size_t)k0 * L + l0, L, tid);
    __syncthreads();
#pragma unroll
    for (int mi = 0; mi < 2; ++mi) {
      v16bf a = AsF[(wm + mi) * 32 + lane];
#pragma unroll
      for (int ni = 0; ni < 2; ++ni) {
        v16bf bb = BsF[(wn + ni) * 32 + lane];
        acc[mi][ni] = wmma_bf16(a, bb, acc[mi][ni]);
      }
    }
    __syncthreads();
  }

  int n = lane & 15, hi = lane >> 4;
#pragma unroll
  for (int mi = 0; mi < 2; ++mi)
#pragma unroll
    for (int ni = 0; ni < 2; ++ni)
#pragma unroll
      for (int r = 0; r < 8; ++r) {
        int e  = e0 + (wm + mi) * 16 + r + 8 * hi;
        int li = l0 + (wn + ni) * 16 + n;
        __bf16 v = f2bf(acc[mi][ni][r]);
        if (which) Qb[((size_t)bh * E + e) * L + li] = v;
        else       Kt[((size_t)bh * L + li) * E + e] = v;
      }
}

// ---------------------------------------------------------------------------
// Kernel 3: softmax row stats (online max / sum-exp), S[i,j] = (Kt Q)/16.
// ---------------------------------------------------------------------------
__global__ __launch_bounds__(256)
void stats_kernel(const __bf16* __restrict__ Kt,
                  const __bf16* __restrict__ Qb,
                  float* __restrict__ Mrow, float* __restrict__ Drow) {
  int bh = blockIdx.y;
  int i0 = blockIdx.x * 64;
  const __bf16* KtBH = Kt + (size_t)bh * L * E;
  const __bf16* QBH  = Qb + (size_t)bh * E * L;

  __shared__ v16bf AsF[4 * 32];     //  4 KB
  __shared__ v16bf BsF[8 * 32];     //  8 KB
  __shared__ float Sbuf[64 * 128];  // 32 KB

  int tid = threadIdx.x, lane = tid & 31, wid = tid >> 5;
  int wm = (wid >> 2) * 2, wn = (wid & 3) * 2;

  float mrun = -3.0e38f, drun = 0.0f;

  for (int j0 = 0; j0 < L; j0 += 128) {
    v8f acc[2][2];
#pragma unroll
    for (int i = 0; i < 2; ++i)
#pragma unroll
      for (int j = 0; j < 2; ++j) zero8(acc[i][j]);

    for (int e0 = 0; e0 < E; e0 += 32) {
      stage_a_bf16<64>(AsF, KtBH + (size_t)i0 * E + e0, E, tid);
      stage_b_bf16<128, 6>(BsF, QBH + (size_t)e0 * L + j0, L, tid);
      wait_async();
      __syncthreads();
#pragma unroll
      for (int mi = 0; mi < 2; ++mi) {
        v16bf a = AsF[(wm + mi) * 32 + lane];
#pragma unroll
        for (int ni = 0; ni < 2; ++ni) {
          v16bf bb = BsF[(wn + ni) * 32 + lane];
          acc[mi][ni] = wmma_bf16(a, bb, acc[mi][ni]);
        }
      }
      __syncthreads();
    }

    int n = lane & 15, hi = lane >> 4;
#pragma unroll
    for (int mi = 0; mi < 2; ++mi)
#pragma unroll
      for (int ni = 0; ni < 2; ++ni)
#pragma unroll
        for (int r = 0; r < 8; ++r)
          Sbuf[((wm + mi) * 16 + r + 8 * hi) * 128 + (wn + ni) * 16 + n] =
              acc[mi][ni][r];
    __syncthreads();

    if (tid < 64) {
      const float4* row = (const float4*)&Sbuf[tid * 128];
      float tmax = -3.0e38f;
      for (int c2 = 0; c2 < 32; ++c2) {
        float4 v = row[c2];
        tmax = fmaxf(tmax, fmaxf(fmaxf(v.x, v.y), fmaxf(v.z, v.w)));
      }
      tmax *= INV_SQRT_C;
      float nm = fmaxf(mrun, tmax);
      float sum = 0.0f;
      for (int c2 = 0; c2 < 32; ++c2) {
        float4 v = row[c2];
        sum += __expf(v.x * INV_SQRT_C - nm) + __expf(v.y * INV_SQRT_C - nm) +
               __expf(v.z * INV_SQRT_C - nm) + __expf(v.w * INV_SQRT_C - nm);
      }
      drun = drun * __expf(mrun - nm) + sum;
      mrun = nm;
    }
    __syncthreads();
  }
  if (tid < 64) {
    Mrow[(size_t)bh * L + i0 + tid] = mrun;
    Drow[(size_t)bh * L + i0 + tid] = drun;
  }
}

// ---------------------------------------------------------------------------
// Kernel 4: fused attention-value (flash-style pass 2).
// ---------------------------------------------------------------------------
__global__ __launch_bounds__(256)
void attnv_kernel(const __bf16* __restrict__ Kt,
                  const __bf16* __restrict__ Qb,
                  const __bf16* __restrict__ xbf,
                  const float* __restrict__ Mrow,
                  const float* __restrict__ Drow,
                  __bf16* __restrict__ Vcat) {  // [B][H][C][L]
  int bh = blockIdx.y;
  int b = bh / H;
  int j0 = blockIdx.x * 64;
  const __bf16* KtBH = Kt + (size_t)bh * L * E;
  const __bf16* QBH  = Qb + (size_t)bh * E * L;
  const __bf16* Xb   = xbf + (size_t)b * C * L;
  const float*  Mr   = Mrow + (size_t)bh * L;
  const float*  Dr   = Drow + (size_t)bh * L;

  __shared__ v16bf QsF[8 * 4 * 32];  // Q[:, j-block] frag-major  (32 KB)
  __shared__ v16bf KsF[8 * 2 * 32];  // Kt[i-slice][:] frag-major (16 KB)
  __shared__ v16bf PbF[4 * 32];      // P tile frag-major          (4 KB)
  __shared__ float mloc[32], dloc[32];

  int tid = threadIdx.x, lane = tid & 31, wid = tid >> 5;
  int smt = (wid >> 2);
  int snt = (wid & 3);
  int t0  = wid * 32;

  for (int es = 0; es < 8; ++es)
    stage_b_bf16<64, 5>(QsF + es * 4 * 32, QBH + (size_t)es * 32 * L + j0, L,
                        tid);

  v8f vacc[2][4];
#pragma unroll
  for (int ti = 0; ti < 2; ++ti)
#pragma unroll
    for (int ji = 0; ji < 4; ++ji) zero8(vacc[ti][ji]);

  for (int i0 = 0; i0 < L; i0 += 32) {
    __syncthreads();  // prev phase-2 done with PbF/KsF; first iter: QsF ready
    for (int es = 0; es < 8; ++es)
      stage_a_bf16<32>(KsF + es * 2 * 32, KtBH + (size_t)i0 * E + es * 32, E,
                       tid);
    if (tid < 32) { mloc[tid] = Mr[i0 + tid]; dloc[tid] = Dr[i0 + tid]; }
    if (i0 + 32 < L)  // prefetch next Kt slice (global_prefetch_b8)
      __builtin_prefetch(KtBH + (size_t)(i0 + 32) * E + tid * 32, 0, 1);
    wait_async();
    __syncthreads();

    // phase 1: S tile (32 x 64); each wave one 16x16 tile, e-contraction 256
    v8f sacc; zero8(sacc);
#pragma unroll
    for (int es = 0; es < 8; ++es) {
      v16bf a  = KsF[(es * 2 + smt) * 32 + lane];
      v16bf bq = QsF[(es * 4 + snt) * 32 + lane];
      sacc = wmma_bf16(a, bq, sacc);
    }
    // normalize -> P, written directly in B-fragment-major layout
    {
      __bf16* Pd = (__bf16*)PbF;
      int n = lane & 15, hi = lane >> 4;
#pragma unroll
      for (int r = 0; r < 8; ++r) {
        int irow = smt * 16 + r + 8 * hi;
        float p = __expf(sacc[r] * INV_SQRT_C - mloc[irow]) / dloc[irow];
        Pd[(snt * 32 + n + 16 * ((irow >> 3) & 1)) * 16 + (irow & 7) + ((irow >> 4) << 3)] = f2bf(p);
      }
    }
    __syncthreads();

    // phase 2: V[t0..t0+31, j-block] += x[t, i-step] @ P
#pragma unroll
    for (int ti = 0; ti < 2; ++ti) {
      v16bf a = load_a_frag_g(Xb + (size_t)(t0 + ti * 16) * L + i0, L, lane);
#pragma unroll
      for (int ji = 0; ji < 4; ++ji)
        vacc[ti][ji] = wmma_bf16(a, PbF[ji * 32 + lane], vacc[ti][ji]);
    }
  }

  __bf16* Vb = Vcat + (size_t)bh * C * L;
  int n = lane & 15, hi = lane >> 4;
#pragma unroll
  for (int ti = 0; ti < 2; ++ti)
#pragma unroll
    for (int ji = 0; ji < 4; ++ji)
#pragma unroll
      for (int r = 0; r < 8; ++r) {
        int t = t0 + ti * 16 + r + 8 * hi;
        int j = j0 + ji * 16 + n;
        Vb[(size_t)t * L + j] = f2bf(vacc[ti][ji][r]);
      }
}

// ---------------------------------------------------------------------------
// Kernel 5: out[b] = Ut (256 x 2048) @ Vcat[b] (2048 x 2048), fp32 out.
// ---------------------------------------------------------------------------
__global__ __launch_bounds__(256)
void unify_kernel(const __bf16* __restrict__ Ut,
                  const __bf16* __restrict__ Vcat,
                  float* __restrict__ out) {
  int b = blockIdx.z;
  int m0 = blockIdx.y * 64;
  int l0 = blockIdx.x * 128;
  const __bf16* Vb = Vcat + (size_t)b * (H * C) * L;
  constexpr int KK = H * C;

  __shared__ v16bf AsF[4 * 32];
  __shared__ v16bf BsF[8 * 32];

  int tid = threadIdx.x, lane = tid & 31, wid = tid >> 5;
  int wm = (wid >> 2) * 2, wn = (wid & 3) * 2;

  v8f acc[2][2];
#pragma unroll
  for (int i = 0; i < 2; ++i)
#pragma unroll
    for (int j = 0; j < 2; ++j) zero8(acc[i][j]);

  for (int k0 = 0; k0 < KK; k0 += 32) {
    stage_a_bf16<64>(AsF, Ut + (size_t)m0 * KK + k0, KK, tid);
    stage_b_bf16<128, 6>(BsF, Vb + (size_t)k0 * L + l0, L, tid);
    wait_async();
    __syncthreads();
#pragma unroll
    for (int mi = 0; mi < 2; ++mi) {
      v16bf a = AsF[(wm + mi) * 32 + lane];
#pragma unroll
      for (int ni = 0; ni < 2; ++ni) {
        v16bf bb = BsF[(wn + ni) * 32 + lane];
        acc[mi][ni] = wmma_bf16(a, bb, acc[mi][ni]);
      }
    }
    __syncthreads();
  }

  int n = lane & 15, hi = lane >> 4;
#pragma unroll
  for (int mi = 0; mi < 2; ++mi)
#pragma unroll
    for (int ni = 0; ni < 2; ++ni)
#pragma unroll
      for (int r = 0; r < 8; ++r) {
        int m = m0 + (wm + mi) * 16 + r + 8 * hi;
        int l = l0 + (wn + ni) * 16 + n;
        out[((size_t)b * C + m) * L + l] = acc[mi][ni][r];
      }
}

// ---------------------------------------------------------------------------
extern "C" void kernel_launch(void* const* d_in, const int* in_sizes, int n_in,
                              void* d_out, int out_size, void* d_ws,
                              size_t ws_size, hipStream_t stream) {
  (void)in_sizes; (void)n_in; (void)out_size; (void)ws_size;
  const float* x       = (const float*)d_in[0];
  const float* tokey   = (const float*)d_in[1];
  const float* toquery = (const float*)d_in[2];
  const float* unify   = (const float*)d_in[3];
  float* out = (float*)d_out;

  char* p = (char*)d_ws;
  auto take = [&](size_t bytes) -> char* {
    char* r = p;
    p += (bytes + 255) & ~(size_t)255;
    return r;
  };
  __bf16* Kt   = (__bf16*)take((size_t)B * H * L * E * 2);  // 32 MB
  __bf16* Qb   = (__bf16*)take((size_t)B * H * E * L * 2);  // 32 MB
  __bf16* xbf  = (__bf16*)take((size_t)B * C * L * 2);      //  4 MB
  __bf16* Ut   = (__bf16*)take((size_t)C * H * C * 2);      //  1 MB
  float*  Mrow = (float*) take((size_t)B * H * L * 4);
  float*  Drow = (float*) take((size_t)B * H * L * 4);
  __bf16* Vcat = (__bf16*)take((size_t)B * H * C * L * 2);  // 32 MB

  prep_kernel<<<2048, 256, 0, stream>>>(x, unify, xbf, Ut);

  dim3 g1(L / 128, E / 64, B * H * 2);
  kq_gemm_kernel<<<g1, 256, 0, stream>>>(tokey, toquery, xbf, Kt, Qb);

  dim3 g2(L / 64, B * H);
  stats_kernel<<<g2, 256, 0, stream>>>(Kt, Qb, Mrow, Drow);
  attnv_kernel<<<g2, 256, 0, stream>>>(Kt, Qb, xbf, Mrow, Drow, Vcat);

  dim3 g3(L / 128, C / 64, B);
  unify_kernel<<<g3, 256, 0, stream>>>(Ut, Vcat, out);
}
